// graph_agg_layer_77197742178845
// MI455X (gfx1250) — compile-verified, matrix operand order checked by reading, not run on hardware
//
#include <hip/hip_runtime.h>
#include <math.h>

#define E_EDGES    327680
#define D_HID      256
#define C_CH       256
#define OUT_DIM    128
#define NUM_GRAPHS 4096
#define EPS_BN     1e-5f

typedef __bf16        v16bf __attribute__((ext_vector_type(16)));
typedef float         v8f   __attribute__((ext_vector_type(8)));
typedef unsigned int  v4u   __attribute__((ext_vector_type(4)));
typedef unsigned int  v8u   __attribute__((ext_vector_type(8)));

// ---------------- init: zero pooled accumulator + stats accumulators ----------------
__global__ void k_zero(float* __restrict__ g, float* __restrict__ acc) {
    int i = blockIdx.x * 256 + threadIdx.x;
    if (i < NUM_GRAPHS * D_HID) g[i] = 0.0f;
    if (i < 512)                acc[i] = 0.0f;
}

// ---------------- f32 -> bf16 one-pass conversion ----------------
__global__ void k_cvt_bf16(const float* __restrict__ src, __bf16* __restrict__ dst) {
    int i = blockIdx.x * 256 + threadIdx.x;
    dst[i] = (__bf16)src[i];
}

// ---------------- scatter-sum: g[batch[src[e]]] += edge[e] ----------------
// One thread per (edge, 4-channel group): float4 load + 4 L2 atomics.
// This is the HBM-bound phase (~335 MB streamed).
__global__ void k_scatter(const float4* __restrict__ edge4,
                          const int*    __restrict__ batch,
                          const int*    __restrict__ src,   // edge_index row 0
                          float*        __restrict__ g) {
    int idx = blockIdx.x * 256 + threadIdx.x;   // E * 64 threads
    int e   = idx >> 6;
    int c4  = idx & 63;
    // hint the streaming read a few blocks ahead
    int pf = idx + 16384;
    if (pf < E_EDGES * 64) __builtin_prefetch(&edge4[pf], 0, 1);
    int gid = batch[src[e]];
    float4 v = edge4[(size_t)e * 64 + c4];
    float* dst = g + (size_t)gid * D_HID + c4 * 4;
    atomicAdd(dst + 0, v.x);
    atomicAdd(dst + 1, v.y);
    atomicAdd(dst + 2, v.z);
    atomicAdd(dst + 3, v.w);
}

// Load one 16x32 bf16 fragment for this lane: row base p (= &M[row*K + k0]),
// lane-half h. Layout (ISA 16-bit A 16x32): VGPR v<4 -> k = h*8 + 2v + {0,1},
// VGPR v>=4 -> k = 16 + h*8 + 2(v-4) + {0,1}. Each half is 8 contiguous bf16
// -> two 16-byte loads per fragment.
static __device__ __forceinline__ v16bf load_frag(const __bf16* __restrict__ p, int h) {
    const v4u lo = *reinterpret_cast<const v4u*>(p + h * 8);
    const v4u hi = *reinterpret_cast<const v4u*>(p + 16 + h * 8);
    v8u r;
    r[0] = lo[0]; r[1] = lo[1]; r[2] = lo[2]; r[3] = lo[3];
    r[4] = hi[0]; r[5] = hi[1]; r[6] = hi[2]; r[7] = hi[3];
    return __builtin_bit_cast(v16bf, r);
}

// ---------------- WMMA GEMM: Y[M,N] = X[M,K] @ W[N,K]^T ----------------
// bf16 inputs, f32 accumulate. Each wave: 4 M-tiles x 1 N-tile (B fragment
// reused 4x). K = 256 fully unrolled -> 32 v_wmma per wave.
template <bool OUT_BF16>
__global__ void k_gemm_bf16(const __bf16* __restrict__ X,
                            const __bf16* __restrict__ W,
                            void*         __restrict__ Yv,
                            int N) {
    constexpr int K = 256;
    const int lane  = threadIdx.x & 31;
    const int wave  = threadIdx.x >> 5;
    const int tile  = blockIdx.x * 8 + wave;
    const int ntile = N >> 4;
    const int tn    = tile % ntile;
    const int mb    = (tile / ntile) * 64;   // 4 M-tiles of 16 rows
    const int l16   = lane & 15;
    const int h     = lane >> 4;
    const int ncol  = (tn << 4) + l16;       // B column (= W row)

    const __bf16* __restrict__ wrow = W + (size_t)ncol * K;
    const __bf16* __restrict__ x0   = X + (size_t)(mb + l16) * K;          // tile 0 row
    const size_t  tstride = (size_t)16 * K;                                 // 16 rows

    v8f c0 = {}, c1 = {}, c2 = {}, c3 = {};
#pragma unroll
    for (int k0 = 0; k0 < K; k0 += 32) {
        const v16bf b  = load_frag(wrow + k0, h);
        const v16bf a0 = load_frag(x0 + k0, h);
        const v16bf a1 = load_frag(x0 + tstride + k0, h);
        const v16bf a2 = load_frag(x0 + 2 * tstride + k0, h);
        const v16bf a3 = load_frag(x0 + 3 * tstride + k0, h);
        c0 = __builtin_amdgcn_wmma_f32_16x16x32_bf16(false, a0, false, b, (short)0, c0, false, false);
        c1 = __builtin_amdgcn_wmma_f32_16x16x32_bf16(false, a1, false, b, (short)0, c1, false, false);
        c2 = __builtin_amdgcn_wmma_f32_16x16x32_bf16(false, a2, false, b, (short)0, c2, false, false);
        c3 = __builtin_amdgcn_wmma_f32_16x16x32_bf16(false, a3, false, b, (short)0, c3, false, false);
    }

    // C/D 16x16 f32 layout: VGPR r -> M = r (lanes 0-15), r+8 (lanes 16-31)
#pragma unroll
    for (int t = 0; t < 4; ++t) {
        const v8f c = (t == 0) ? c0 : (t == 1) ? c1 : (t == 2) ? c2 : c3;
#pragma unroll
        for (int r = 0; r < 8; ++r) {
            size_t idx = (size_t)(mb + t * 16 + r + 8 * h) * N + ncol;
            if (OUT_BF16) ((__bf16*)Yv)[idx] = (__bf16)c[r];
            else          ((float*)Yv)[idx]  = c[r];
        }
    }
}

// ---------------- per-channel batch stats: acc=sum(x), acc2=sum(x^2) ----------------
__global__ void k_stats(const float* __restrict__ g2,
                        float* __restrict__ acc, float* __restrict__ acc2) {
    int c  = threadIdx.x;
    int r0 = blockIdx.x * 16;
    float s = 0.0f, s2 = 0.0f;
#pragma unroll
    for (int i = 0; i < 16; ++i) {
        float x = g2[(size_t)(r0 + i) * C_CH + c];
        s  += x;
        s2 += x * x;
    }
    atomicAdd(&acc[c],  s);
    atomicAdd(&acc2[c], s2);
}

// ---------------- BatchNorm (biased stats) + exact-erf GELU -> bf16 ----------------
__global__ void k_bn_gelu(const float* __restrict__ g2,
                          const float* __restrict__ acc, const float* __restrict__ acc2,
                          const float* __restrict__ gamma, const float* __restrict__ beta,
                          __bf16* __restrict__ g3) {
    int c = threadIdx.x;
    int r = blockIdx.x;
    const float invG = 1.0f / (float)NUM_GRAPHS;
    float mean = acc[c] * invG;
    float var  = acc2[c] * invG - mean * mean;
    float inv  = rsqrtf(var + EPS_BN);
    float x = (g2[(size_t)r * C_CH + c] - mean) * inv * gamma[c] + beta[c];
    float y = 0.5f * x * (1.0f + erff(x * 0.70710678118654752440f));
    g3[(size_t)r * C_CH + c] = (__bf16)y;
}

extern "C" void kernel_launch(void* const* d_in, const int* in_sizes, int n_in,
                              void* d_out, int out_size, void* d_ws, size_t ws_size,
                              hipStream_t stream) {
    const float* edge  = (const float*)d_in[0];   // [E, 256]
    const int*   batch = (const int*)  d_in[1];   // [N_NODES]
    const int*   eidx  = (const int*)  d_in[2];   // [2, E]; row 0 = sources
    const float* W1    = (const float*)d_in[3];   // [256, 256]
    const float* W2    = (const float*)d_in[4];   // [256, 256]
    const float* W3    = (const float*)d_in[5];   // [128, 256]
    const float* gamma = (const float*)d_in[6];   // [256]
    const float* beta  = (const float*)d_in[7];   // [256]
    float*       out   = (float*)d_out;           // [4096, 128]

    // ---- workspace layout ----
    char* ws = (char*)d_ws;
    float*  bufF = (float*)ws;                               // 4 MB: pooled g, then g2
    __bf16* aB   = (__bf16*)(ws + (4u << 20));               // 2 MB: pooled g bf16
    __bf16* g1B  = (__bf16*)(ws + (6u << 20));               // 2 MB: GEMM1 out bf16
    __bf16* g3B  = (__bf16*)(ws + (8u << 20));               // 2 MB: BN/GELU out bf16
    __bf16* W1B  = (__bf16*)(ws + (10u << 20));              // 128 KB
    __bf16* W2B  = W1B + 256 * 256;                          // 128 KB
    __bf16* W3B  = W2B + 256 * 256;                          // 64 KB
    float*  acc  = (float*)(W3B + 128 * 256);                // 256 sums
    float*  acc2 = acc + 256;                                // 256 sum-of-squares

    // 1) zero pooled accumulator + stats
    k_zero<<<(NUM_GRAPHS * D_HID) / 256, 256, 0, stream>>>(bufF, acc);

    // 2) one-pass weight conversion to bf16
    k_cvt_bf16<<<(256 * 256) / 256, 256, 0, stream>>>(W1, W1B);
    k_cvt_bf16<<<(256 * 256) / 256, 256, 0, stream>>>(W2, W2B);
    k_cvt_bf16<<<(128 * 256) / 256, 256, 0, stream>>>(W3, W3B);

    // 3) scatter-sum edges into per-graph rows (HBM-bound)
    k_scatter<<<(E_EDGES * (D_HID / 4)) / 256, 256, 0, stream>>>(
        (const float4*)edge, batch, eidx, bufF);

    // 4) pooled g -> bf16
    k_cvt_bf16<<<(NUM_GRAPHS * D_HID) / 256, 256, 0, stream>>>(bufF, aB);

    // 5) g @ W1^T -> g1 (bf16): 64 row-groups x 16 col-tiles = 1024 waves
    k_gemm_bf16<true><<<(NUM_GRAPHS / 64) * (D_HID / 16) / 8, 256, 0, stream>>>(
        aB, W1B, (void*)g1B, D_HID);

    // 6) g1 @ W2^T -> g2 (f32, feeds statistics)
    k_gemm_bf16<false><<<(NUM_GRAPHS / 64) * (C_CH / 16) / 8, 256, 0, stream>>>(
        g1B, W2B, (void*)bufF, C_CH);

    // 7) batch statistics
    k_stats<<<NUM_GRAPHS / 16, C_CH, 0, stream>>>(bufF, acc, acc2);

    // 8) BN + exact GELU -> bf16
    k_bn_gelu<<<NUM_GRAPHS, C_CH, 0, stream>>>(bufF, acc, acc2, gamma, beta, g3B);

    // 9) g3 @ W3^T -> out (f32)
    k_gemm_bf16<false><<<(NUM_GRAPHS / 64) * (OUT_DIM / 16) / 8, 256, 0, stream>>>(
        g3B, W3B, (void*)out, OUT_DIM);
}
